// BertGGCN_38130719654091
// MI455X (gfx1250) — matrix operand order repaired and checked by reading, not compiled, not run on hardware
//
#include <hip/hip_runtime.h>
#include <hip/hip_bf16.h>
#include <cstddef>
#include <cstdint>

// ---------------- Types for CDNA5 WMMA ----------------
typedef __attribute__((ext_vector_type(8)))  __bf16 v8bf;
typedef __attribute__((ext_vector_type(16))) __bf16 v16bf;
typedef __attribute__((ext_vector_type(8)))  float  v8f;
typedef unsigned short u16;

// ---------------- Problem constants ----------------
#define NNODES 51200
#define NEDGES 409600
#define FEAT   768
#define HID    256
#define LAYERS 6
#define NPG    200
#define NGRAPH 256          // NNODES / NPG
#define C1_OUT 50
#define C2_OUT 20
#define CHUNK  12800        // GRU node chunk (4 chunks)

// fp32 -> bf16, round-to-nearest-even
__device__ __forceinline__ u16 f2bf(float f) {
    union { float f; unsigned u; } v; v.f = f;
    unsigned r = v.u + 0x7FFFu + ((v.u >> 16) & 1u);
    return (u16)(r >> 16);
}

__device__ __forceinline__ v16bf ld_frag(const u16* lo_p, const u16* hi_p) {
    v8bf lo = *reinterpret_cast<const v8bf*>(lo_p);
    v8bf hi = *reinterpret_cast<const v8bf*>(hi_p);
    return __builtin_shufflevector(lo, hi, 0,1,2,3,4,5,6,7,8,9,10,11,12,13,14,15);
}

// CDNA5 async copy: 16 bytes global -> LDS, tracked by ASYNCcnt
__device__ __forceinline__ void async_copy_b128(uint32_t lds_off, uint64_t gaddr) {
    asm volatile("global_load_async_to_lds_b128 %0, %1, off"
                 :: "v"(lds_off), "v"(gaddr) : "memory");
}

// =====================================================================
// fp32 -> bf16 conversion (packed pair stores); n must be even
// =====================================================================
__global__ __launch_bounds__(256)
void f32_to_bf16_kernel(const float* __restrict__ in, u16* __restrict__ out, long n) {
    long i = ((long)blockIdx.x * 256 + threadIdx.x) * 2;
    if (i >= n) return;
    unsigned p = (unsigned)f2bf(in[i]) | ((unsigned)f2bf(in[i + 1]) << 16);
    *reinterpret_cast<unsigned*>(out + i) = p;
}

// fp32 [K,N] -> bf16 [N,K]  (for W_ggnn layers)
__global__ __launch_bounds__(256)
void transpose_bf16_kernel(const float* __restrict__ in, u16* __restrict__ out,
                           int K, int N) {
    int idx = blockIdx.x * 256 + threadIdx.x;
    if (idx >= K * N) return;
    int k = idx / N, n = idx % N;
    out[(size_t)n * K + k] = f2bf(in[idx]);
}

// =====================================================================
// WMMA bf16 GEMM: C[M,N] = A[M,K] * B^T + bias, A:[M,K] bf16, B:[N,K] bf16
// Block = 256 threads = 8 waves; block tile 128(M) x 128(N);
// wave tile 32(M) x 64(N) = 2x4 WMMA frags; K-step 32.
// LDS double-buffered; tiles staged with global_load_async_to_lds_b128;
// next tile's DMA overlaps current tile's WMMAs (in-order ASYNCcnt <= 4).
// Requires M%128==0, N%128==0, K%32==0 (true for every call here).
// bias must be a valid pointer (use a zero buffer for "no bias").
// =====================================================================
__global__ __launch_bounds__(256)
void gemm_wmma_bf16t(const u16* __restrict__ A, int lda,
                     const u16* __restrict__ B, int ldb,
                     const float* __restrict__ bias,
                     float* __restrict__ C, int ldc,
                     int M, int N, int K) {
    __shared__ __align__(16) u16 sA[2][128 * 40]; // [m][k], pad 8 (row pitch 80 B)
    __shared__ __align__(16) u16 sB[2][128 * 40]; // [n][k], pad 8

    const int tid  = threadIdx.x;
    const int lane = tid & 31;
    const int wave = tid >> 5;
    const int wm   = wave & 3;   // wave M index (0..3)
    const int wn   = wave >> 2;  // wave N index (0..1)
    const int m0   = blockIdx.y * 128;
    const int n0   = blockIdx.x * 128;
    const int lm   = lane & 15;
    const int half = lane >> 4;

    // per-thread staging chunks: idx -> row = idx>>2, col8 = (idx&3)*8
    const int r0 = tid >> 2, r1 = (256 + tid) >> 2;
    const int c0 = (tid & 3) << 3;
    const u16* gA0 = A + (size_t)(m0 + r0) * lda + c0;
    const u16* gA1 = A + (size_t)(m0 + r1) * lda + c0;
    const u16* gB0 = B + (size_t)(n0 + r0) * ldb + c0;
    const u16* gB1 = B + (size_t)(n0 + r1) * ldb + c0;

    auto stage = [&](int k0, int buf) {
        async_copy_b128((uint32_t)(uintptr_t)&sA[buf][r0 * 40 + c0], (uint64_t)(uintptr_t)(gA0 + k0));
        async_copy_b128((uint32_t)(uintptr_t)&sA[buf][r1 * 40 + c0], (uint64_t)(uintptr_t)(gA1 + k0));
        async_copy_b128((uint32_t)(uintptr_t)&sB[buf][r0 * 40 + c0], (uint64_t)(uintptr_t)(gB0 + k0));
        async_copy_b128((uint32_t)(uintptr_t)&sB[buf][r1 * 40 + c0], (uint64_t)(uintptr_t)(gB1 + k0));
    };

    v8f acc[2][4] = {};
    const int steps = K >> 5;

    stage(0, 0);                                   // prologue: tile 0 -> buf 0
    for (int s = 0; s < steps; ++s) {
        const int buf = s & 1;
        if (s + 1 < steps) {
            stage((s + 1) << 5, buf ^ 1);          // DMA next tile under compute
            asm volatile("s_wait_asynccnt 4" ::: "memory");  // tile s complete
        } else {
            asm volatile("s_wait_asynccnt 0" ::: "memory");
        }
        __syncthreads();                           // tile s visible to all waves

        // ---- fragments per ISA 16-bit layouts ----
        v16bf afr[2], bfr[4];
        #pragma unroll
        for (int t = 0; t < 2; ++t) {
            int row = wm * 32 + t * 16 + lm;
            // A 16x32: elems 0..7 -> K = 8*half+e ; elems 8..15 -> K = 16+8*half+e
            afr[t] = ld_frag(&sA[buf][row * 40 + 8 * half],
                             &sA[buf][row * 40 + 16 + 8 * half]);
        }
        #pragma unroll
        for (int u = 0; u < 4; ++u) {
            int n = wn * 64 + u * 16 + lm;
            // B 32x16: lane n = lane&15, elems 0..7 -> K = 8*half+e ; elems 8..15 -> K = 16+8*half+e
            bfr[u] = ld_frag(&sB[buf][n * 40 + 8 * half],
                             &sB[buf][n * 40 + 16 + 8 * half]);
        }
        #pragma unroll
        for (int t = 0; t < 2; ++t)
            #pragma unroll
            for (int u = 0; u < 4; ++u)
                acc[t][u] = __builtin_amdgcn_wmma_f32_16x16x32_bf16(
                    false, afr[t], false, bfr[u], (short)0, acc[t][u], false, false);
        __syncthreads();   // all waves done reading buf before it is re-filled
    }

    // ---- epilogue: C/D layout (VGPR r -> M = 8*half + r, lane -> N) ----
    #pragma unroll
    for (int u = 0; u < 4; ++u) {
        int gn = n0 + wn * 64 + u * 16 + lm;
        float bv = bias[gn];
        #pragma unroll
        for (int t = 0; t < 2; ++t) {
            float* Cp = C + (size_t)(m0 + wm * 32 + t * 16 + half * 8) * ldc + gn;
            #pragma unroll
            for (int r = 0; r < 8; ++r) {
                Cp[0] = acc[t][u][r] + bv;
                Cp += ldc;
            }
        }
    }
}

// =====================================================================
// segment_sum: agg[dst[e]] += m[src[e]]    (one block per edge, f per lane)
// =====================================================================
__global__ __launch_bounds__(256)
void scatter_add_kernel(const float* __restrict__ m,
                        const int* __restrict__ src,
                        const int* __restrict__ dst,
                        float* __restrict__ agg) {
    int e = blockIdx.x;
    int f = threadIdx.x;
    int s = src[e], d = dst[e];
    float v = m[(size_t)s * HID + f];
    unsafeAtomicAdd(&agg[(size_t)d * HID + f], v);  // global_atomic_add_f32
}

// =====================================================================
// GRUCell update (in place on h chunk); gi/gh are [rows, 3*HID] with bias added
// =====================================================================
__global__ __launch_bounds__(256)
void gru_kernel(float* __restrict__ h,
                const float* __restrict__ gi,
                const float* __restrict__ gh, int rows) {
    int idx = blockIdx.x * 256 + threadIdx.x;
    if (idx >= rows * HID) return;
    int n = idx >> 8, f = idx & 255;
    const float* gin = gi + (size_t)n * 768;
    const float* ghn = gh + (size_t)n * 768;
    float r  = 1.0f / (1.0f + expf(-(gin[f]       + ghn[f])));
    float z  = 1.0f / (1.0f + expf(-(gin[256 + f] + ghn[256 + f])));
    float nn = tanhf(gin[512 + f] + r * ghn[512 + f]);
    float hv = h[idx];
    h[idx] = (1.0f - z) * nn + z * hv;
}

// =====================================================================
// Conv head
// =====================================================================
__device__ __forceinline__ float in_elem(const float* __restrict__ h,
                                         const float* __restrict__ x,
                                         int node, int pos, int Lin, int mode) {
    if (pos < 0 || pos >= Lin) return 0.0f;
    if (mode == 0)
        return (pos < HID) ? h[(size_t)node * HID + pos]
                           : x[(size_t)node * FEAT + (pos - HID)];
    return h[(size_t)node * HID + pos];
}

// conv1d (in_ch=200, K=3, pad=1) + ReLU. mode0: concat(h,x) [*,1024]; mode1: h [*,256]
__global__ __launch_bounds__(256)
void conv1_relu_kernel(const float* __restrict__ h, const float* __restrict__ x,
                       const float* __restrict__ Wc1, const float* __restrict__ bc1,
                       float* __restrict__ out, int Lin, int mode) {
    __shared__ float w[NPG * 3];
    int co = blockIdx.y, g = blockIdx.z;
    for (int i = threadIdx.x; i < NPG * 3; i += 256) w[i] = Wc1[co * NPG * 3 + i];
    __syncthreads();
    int l = blockIdx.x * 256 + threadIdx.x;
    if (l >= Lin) return;
    float acc = bc1[co];
    int nodeBase = g * NPG;
    for (int ci = 0; ci < NPG; ++ci) {
        int node = nodeBase + ci;
        float v0 = in_elem(h, x, node, l - 1, Lin, mode);
        float v1 = in_elem(h, x, node, l,     Lin, mode);
        float v2 = in_elem(h, x, node, l + 1, Lin, mode);
        acc = fmaf(w[ci * 3 + 0], v0, acc);
        acc = fmaf(w[ci * 3 + 1], v1, acc);
        acc = fmaf(w[ci * 3 + 2], v2, acc);
    }
    out[((size_t)g * C1_OUT + co) * Lin + l] = fmaxf(acc, 0.0f);
}

__global__ __launch_bounds__(256)
void maxpool_kernel(const float* __restrict__ in, float* __restrict__ out,
                    int GC, int Lin, int Lout, int k, int s) {
    int idx = blockIdx.x * 256 + threadIdx.x;
    if (idx >= GC * Lout) return;
    int gc = idx / Lout, lo = idx % Lout;
    const float* p = in + (size_t)gc * Lin + lo * s;
    float m = p[0];
    for (int j = 1; j < k; ++j) m = fmaxf(m, p[j]);
    out[idx] = m;
}

// conv1d K=1: [G,50,L] -> [G,20,L]
__global__ __launch_bounds__(256)
void conv2_kernel(const float* __restrict__ in, const float* __restrict__ Wc2,
                  const float* __restrict__ bc2, float* __restrict__ out, int L) {
    int idx = blockIdx.x * 256 + threadIdx.x;
    if (idx >= NGRAPH * C2_OUT * L) return;
    int l = idx % L; int t = idx / L; int c2 = t % C2_OUT; int g = t / C2_OUT;
    float acc = bc2[c2];
    const float* wr = Wc2 + c2 * C1_OUT;
    const float* ip = in + (size_t)g * C1_OUT * L + l;
    #pragma unroll 5
    for (int c1 = 0; c1 < C1_OUT; ++c1) acc = fmaf(wr[c1], ip[(size_t)c1 * L], acc);
    out[idx] = acc;
}

// fused FC + sigmoid(z*y) + clip + logit; out[g] = {0, z1}
__global__ __launch_bounds__(256)
void head_kernel(const float* __restrict__ Z, const float* __restrict__ Y,
                 const float* __restrict__ Wf1, const float* __restrict__ bf1v,
                 const float* __restrict__ Wf2, const float* __restrict__ bf2v,
                 float* __restrict__ out) {
    __shared__ float rz[256], ry[256];
    int g = blockIdx.x, tid = threadIdx.x;
    float az = 0.0f, ay = 0.0f;
    for (int i = tid; i < 5100; i += 256) az = fmaf(Z[(size_t)g * 5100 + i], Wf1[i], az);
    for (int i = tid; i < 1260; i += 256) ay = fmaf(Y[(size_t)g * 1260 + i], Wf2[i], ay);
    rz[tid] = az; ry[tid] = ay;
    __syncthreads();
    for (int s = 128; s > 0; s >>= 1) {
        if (tid < s) { rz[tid] += rz[tid + s]; ry[tid] += ry[tid + s]; }
        __syncthreads();
    }
    if (tid == 0) {
        float z = rz[0] + bf1v[0];
        float y = ry[0] + bf2v[0];
        float p = 1.0f / (1.0f + expf(-(z * y)));
        const float eps = 1e-6f;
        p = fminf(fmaxf(p, eps), 1.0f - eps);
        out[g * 2 + 0] = 0.0f;
        out[g * 2 + 1] = logf(p / (1.0f - p));
    }
}

// =====================================================================
// Host launch
// =====================================================================
extern "C" void kernel_launch(void* const* d_in, const int* in_sizes, int n_in,
                              void* d_out, int out_size, void* d_ws, size_t ws_size,
                              hipStream_t stream) {
    const float* x    = (const float*)d_in[0];
    const int*   ei   = (const int*)  d_in[1];
    const float* W_in = (const float*)d_in[2];
    const float* b_in = (const float*)d_in[3];
    const float* W_g  = (const float*)d_in[4];
    const float* W_ih = (const float*)d_in[5];
    const float* W_hh = (const float*)d_in[6];
    const float* b_ih = (const float*)d_in[7];
    const float* b_hh = (const float*)d_in[8];
    const float* Wc1  = (const float*)d_in[9];
    const float* bc1  = (const float*)d_in[10];
    const float* Wc2  = (const float*)d_in[11];
    const float* bc2  = (const float*)d_in[12];
    const float* Wf1  = (const float*)d_in[13];
    const float* bf1  = (const float*)d_in[14];
    const float* Wf2  = (const float*)d_in[15];
    const float* bf2  = (const float*)d_in[16];
    float* out = (float*)d_out;

    // ---- workspace layout ----
    // fp32 region (conv head aliases GGNN scratch); then bf16 region.
    float* ws  = (float*)d_ws;
    float* h   = ws;                      // 51200*256 = 13,107,200 f
    float* mb  = h   + 13107200;          // 13,107,200 f
    float* agg = mb  + 13107200;          // 13,107,200 f
    float* gi  = agg + 13107200;          //  9,830,400 f (chunk 12800*768)
    float* gh  = gi  + 9830400;           //  9,830,400 f
    // conv-head aliases (GGNN scratch dead by then)
    float* Z1 = mb;                       // 256*50*1024
    float* Z2 = agg;                      // 256*50*511
    float* Z3 = gi;                       // 256*20*511
    float* Y1 = gi + 2616320;             // 256*50*256
    float* Z4 = gh;                       // 256*20*255
    float* Y2 = gh + 1305600;             // 256*50*127
    float* Y3 = Y2 + 1625600;             // 256*20*127
    float* Y4 = Y3 + 650240;              // 256*20*63
    // bf16 region
    u16* xb    = (u16*)(gh + 9830400);    // 39,321,600
    u16* hb    = xb    + 39321600;        // 13,107,200
    u16* aggb  = hb    + 13107200;        // 13,107,200
    u16* W_inb = aggb  + 13107200;        //    196,608
    u16* WgTb  = W_inb + 196608;          //    393,216 (6 layers, [N,K])
    u16* W_ihb = WgTb  + 393216;          //    196,608
    u16* W_hhb = W_ihb + 196608;          //    196,608
    float* zbias = (float*)(W_hhb + 196608); // 768 f of zeros ("no bias")

    const int* src = ei;
    const int* dst = ei + NEDGES;
    dim3 blk(256);

    // ---- one-time conversions to bf16 + zero-bias ----
    hipMemsetAsync(zbias, 0, 768 * sizeof(float), stream);
    f32_to_bf16_kernel<<<(39321600 / 2 + 255) / 256, blk, 0, stream>>>(x, xb, 39321600);
    f32_to_bf16_kernel<<<(196608 / 2 + 255) / 256, blk, 0, stream>>>(W_in, W_inb, 196608);
    f32_to_bf16_kernel<<<(196608 / 2 + 255) / 256, blk, 0, stream>>>(W_ih, W_ihb, 196608);
    f32_to_bf16_kernel<<<(196608 / 2 + 255) / 256, blk, 0, stream>>>(W_hh, W_hhb, 196608);
    for (int l = 0; l < LAYERS; ++l)
        transpose_bf16_kernel<<<(65536 + 255) / 256, blk, 0, stream>>>(
            W_g + (size_t)l * 65536, WgTb + (size_t)l * 65536, HID, HID);

    // h0 = x @ W_in^T + b_in   [51200,768]x[768,256]
    gemm_wmma_bf16t<<<dim3(HID / 128, NNODES / 128), blk, 0, stream>>>(
        xb, FEAT, W_inb, FEAT, b_in, h, HID, NNODES, HID, FEAT);

    for (int layer = 0; layer < LAYERS; ++layer) {
        // snapshot h -> bf16 (used by both m-GEMM and gh-GEMM of this layer)
        f32_to_bf16_kernel<<<(13107200 / 2 + 255) / 256, blk, 0, stream>>>(h, hb, 13107200);
        // m = h @ W_ggnn[layer]   (WgTb is [N,K])
        gemm_wmma_bf16t<<<dim3(HID / 128, NNODES / 128), blk, 0, stream>>>(
            hb, HID, WgTb + (size_t)layer * 65536, HID, zbias, mb, HID,
            NNODES, HID, HID);
        // agg = segment_sum(m[src], dst)
        hipMemsetAsync(agg, 0, (size_t)NNODES * HID * sizeof(float), stream);
        scatter_add_kernel<<<NEDGES, blk, 0, stream>>>(mb, src, dst, agg);
        f32_to_bf16_kernel<<<(13107200 / 2 + 255) / 256, blk, 0, stream>>>(agg, aggb, 13107200);
        // GRU, chunked over nodes
        for (int c = 0; c < NNODES / CHUNK; ++c) {
            float* hC  = h    + (size_t)c * CHUNK * HID;
            u16*   hbC = hb   + (size_t)c * CHUNK * HID;
            u16*   abC = aggb + (size_t)c * CHUNK * HID;
            gemm_wmma_bf16t<<<dim3(768 / 128, CHUNK / 128), blk, 0, stream>>>(
                abC, HID, W_ihb, HID, b_ih, gi, 768, CHUNK, 768, HID);
            gemm_wmma_bf16t<<<dim3(768 / 128, CHUNK / 128), blk, 0, stream>>>(
                hbC, HID, W_hhb, HID, b_hh, gh, 768, CHUNK, 768, HID);
            gru_kernel<<<(CHUNK * HID) / 256, blk, 0, stream>>>(hC, gi, gh, CHUNK);
        }
    }

    // ---- conv head, Z path (concat(h,x), L=1024 -> 511 -> 255) ----
    conv1_relu_kernel<<<dim3(1024 / 256, C1_OUT, NGRAPH), blk, 0, stream>>>(
        h, x, Wc1, bc1, Z1, 1024, 0);
    maxpool_kernel<<<(NGRAPH * C1_OUT * 511 + 255) / 256, blk, 0, stream>>>(
        Z1, Z2, NGRAPH * C1_OUT, 1024, 511, 3, 2);
    conv2_kernel<<<(NGRAPH * C2_OUT * 511 + 255) / 256, blk, 0, stream>>>(
        Z2, Wc2, bc2, Z3, 511);
    maxpool_kernel<<<(NGRAPH * C2_OUT * 255 + 255) / 256, blk, 0, stream>>>(
        Z3, Z4, NGRAPH * C2_OUT, 511, 255, 2, 2);

    // ---- conv head, Y path (h only, L=256 -> 127 -> 63) ----
    conv1_relu_kernel<<<dim3(1, C1_OUT, NGRAPH), blk, 0, stream>>>(
        h, x, Wc1, bc1, Y1, 256, 1);
    maxpool_kernel<<<(NGRAPH * C1_OUT * 127 + 255) / 256, blk, 0, stream>>>(
        Y1, Y2, NGRAPH * C1_OUT, 256, 127, 3, 2);
    conv2_kernel<<<(NGRAPH * C2_OUT * 127 + 255) / 256, blk, 0, stream>>>(
        Y2, Wc2, bc2, Y3, 127);
    maxpool_kernel<<<(NGRAPH * C2_OUT * 63 + 255) / 256, blk, 0, stream>>>(
        Y3, Y4, NGRAPH * C2_OUT, 127, 63, 2, 2);

    // ---- fused FCs + sigmoid + logit ----
    head_kernel<<<NGRAPH, blk, 0, stream>>>(Z4, Y4, Wf1, bf1, Wf2, bf2, out);
}
